// MultimodalFusion_90649579750015
// MI455X (gfx1250) — matrix operand order repaired and checked by reading, not compiled, not run on hardware
//
#include <hip/hip_runtime.h>

// ---------------- problem constants ----------------
#define BATCH   16384
#define DMODEL  512
#define NHEAD   8
#define HDIM    64
#define FFDIM   2048
#define SEQ     3
#define TB      16              // batch items per workgroup
#define MROWS   (TB * SEQ)      // 48 token rows -> 3 WMMA M-tiles
#define NTHREADS 256
#define NWAVES   8

// ---------------- LDS layout ----------------
#define LXB   520               // bf16 row stride for x / attn-out (pad vs 512)
#define LQKV  1544              // bf16 row stride for qkv
#define LH    264               // bf16 row stride for FF h-chunk (256 + pad)
#define LX2   520               // f32 row stride for pre-LN buffer

#define OFF_XB   0                                   // 48*520*2  = 49,920
#define OFF_QKV  50176                               // 48*1544*2 = 148,224 (reused: H, X2F)
#define OFF_AO   198400                              // 48*520*2  = 49,920
#define OFF_ST   248320                              // stats: 2*48 f32
#define SMEM_BYTES 249856

// ---------------- bf16 fragment-packed weight layout in d_ws ----------------
// A B-fragment (K=32 x N=16) is 32 lanes x 16 bf16 = 1024 bytes, stored
// contiguously; lane l's data at blob + l*32 bytes. frag id = ntile*KT + ktile.
#define FRAG_ELEMS 512          // bf16 elems per fragment blob (1024 B)
#define WS_WIN  0               // w_in  : 96 ntiles * 16 ktiles = 1536 frags (1.5 MB)
#define WS_WOUT 1572864         // w_out : 32 * 16 =  512 frags (0.5 MB)
#define WS_W1   2097152         // w1    : 128 * 16 = 2048 frags (2 MB)
#define WS_W2   4194304         // w2    : 32 * 64 = 2048 frags (2 MB)

typedef __bf16 bf16;
typedef __attribute__((ext_vector_type(16))) bf16  v16bf;
typedef __attribute__((ext_vector_type(8)))  bf16  v8bf;
typedef __attribute__((ext_vector_type(8)))  float v8f;
typedef __attribute__((ext_vector_type(4)))  float v4f;

// ---------------- prepass: f32 [N][K] row-major -> bf16 fragment-major ----------------
__global__ void __launch_bounds__(NTHREADS)
repack_weight(const float* __restrict__ w, bf16* __restrict__ dst, int N, int K)
{
    const int gid  = blockIdx.x * NTHREADS + threadIdx.x;
    const int lane = gid & 31;
    const int frag = gid >> 5;
    const int KT = K >> 5;                     // ktiles per ntile row
    const int nfrag = (N >> 4) * KT;
    if (frag >= nfrag) return;
    const int nt = frag / KT, kt = frag - nt * KT;
    const int n  = nt * 16 + (lane & 15);
    const int kb = kt * 32 + (lane >> 4) * 8;
    const float* p = w + (size_t)n * K + kb;
    bf16* q = dst + (size_t)frag * FRAG_ELEMS + lane * 16;
    v8bf o0, o1;
#pragma unroll
    for (int j = 0; j < 2; ++j) {
        v4f a = *(const v4f*)(p + j * 4);
#pragma unroll
        for (int i = 0; i < 4; ++i) o0[j * 4 + i] = (bf16)a[i];
    }
#pragma unroll
    for (int j = 0; j < 2; ++j) {
        v4f a = *(const v4f*)(p + 16 + j * 4);
#pragma unroll
        for (int i = 0; i < 4; ++i) o1[j * 4 + i] = (bf16)a[i];
    }
    *(v8bf*)(q)     = o0;       // global_store_b128
    *(v8bf*)(q + 8) = o1;       // global_store_b128
}

// ---------------- fragment loaders ----------------
// A fragment (16x32 bf16) from LDS, per-ISA 16-bit A layout:
// lanes 0-15 hold K 0..7 & 16..23, lanes 16-31 hold K 8..15 & 24..31.
__device__ __forceinline__ v16bf load_a(const bf16* rowbase, int lda, int k0, int lane) {
    const int m  = lane & 15;
    const int h2 = lane >> 4;
    const bf16* p = rowbase + m * lda + k0 + h2 * 8;
    v8bf a0 = *(const v8bf*)(p);        // ds_load_b128
    v8bf a1 = *(const v8bf*)(p + 16);   // ds_load_b128
    return __builtin_shufflevector(a0, a1, 0,1,2,3,4,5,6,7,8,9,10,11,12,13,14,15);
}

// B fragment from fragment-packed bf16 weights: two coalesced global_load_b128.
__device__ __forceinline__ v16bf load_b(const bf16* wf, int frag, int lane) {
    const bf16* p = wf + (size_t)frag * FRAG_ELEMS + lane * 16;
    v8bf b0 = *(const v8bf*)(p);
    v8bf b1 = *(const v8bf*)(p + 8);
    return __builtin_shufflevector(b0, b1, 0,1,2,3,4,5,6,7,8,9,10,11,12,13,14,15);
}

__device__ __forceinline__ v8f wmma_bf16(v16bf a, v16bf b, v8f c) {
    return __builtin_amdgcn_wmma_f32_16x16x32_bf16(false, a, false, b, (short)0, c, false, false);
}

__global__ void __launch_bounds__(NTHREADS)
fused_transformer_block(const float* __restrict__ feat0, const float* __restrict__ feat1,
                        const float* __restrict__ feat2, const float* __restrict__ pos_emb,
                        const bf16* __restrict__ win_f,  const float* __restrict__ b_in,
                        const bf16* __restrict__ wout_f, const float* __restrict__ b_out,
                        const float* __restrict__ ln1_g, const float* __restrict__ ln1_b,
                        const bf16* __restrict__ w1_f,   const float* __restrict__ b1,
                        const bf16* __restrict__ w2_f,   const float* __restrict__ b2,
                        const float* __restrict__ ln2_g, const float* __restrict__ ln2_b,
                        float* __restrict__ out)
{
    extern __shared__ char smem[];
    bf16*  XB  = (bf16*)(smem + OFF_XB);    // x, later x1 (post-LN1)
    bf16*  QKV = (bf16*)(smem + OFF_QKV);
    bf16*  H   = (bf16*)(smem + OFF_QKV);   // reuse after attention
    float* X2F = (float*)(smem + OFF_QKV);  // reuse: pre-LN f32 buffer
    bf16*  AO  = (bf16*)(smem + OFF_AO);    // attention output
    float* MU  = (float*)(smem + OFF_ST);
    float* RS  = MU + MROWS;

    const int tid  = threadIdx.x;
    const int lane = tid & 31;
    const int wave = tid >> 5;
    const int b0   = blockIdx.x * TB;
    const int ln15 = lane & 15;
    const int mb8  = (lane >> 4) * 8;

    // ---- 1) x = stack(feat0..2) + pos_emb -> bf16 in LDS ----
    for (int idx = tid; idx < MROWS * DMODEL; idx += NTHREADS) {
        const int r = idx >> 9, c = idx & (DMODEL - 1);
        const int bl = r / SEQ, s = r - bl * SEQ;
        const float* fp = (s == 0) ? feat0 : ((s == 1) ? feat1 : feat2);
        float v = fp[(size_t)(b0 + bl) * DMODEL + c] + pos_emb[s * DMODEL + c];
        XB[r * LXB + c] = (bf16)v;
    }
    __syncthreads();

    // ---- 2) QKV projection: each wave owns an N-column, 3 M-tiles at once ----
    for (int nt = wave; nt < 96; nt += NWAVES) {
        v8f acc[3];
#pragma unroll
        for (int m = 0; m < 3; ++m) acc[m] = (v8f){0.f,0.f,0.f,0.f,0.f,0.f,0.f,0.f};
        for (int kt = 0; kt < DMODEL / 32; ++kt) {
            v16bf bw = load_b(win_f, nt * (DMODEL / 32) + kt, lane);  // 1 B for 3 WMMA
#pragma unroll
            for (int m = 0; m < 3; ++m) {
                v16bf a = load_a(XB + m * 16 * LXB, LXB, kt * 32, lane);
                acc[m] = wmma_bf16(a, bw, acc[m]);
            }
        }
        const int col  = nt * 16 + ln15;
        const float bias = b_in[col];
#pragma unroll
        for (int m = 0; m < 3; ++m) {
            const int mbase = m * 16 + mb8;
#pragma unroll
            for (int r = 0; r < 8; ++r)
                QKV[(mbase + r) * LQKV + col] = (bf16)(acc[m][r] + bias);
        }
    }
    __syncthreads();

    // ---- 3) attention: S=3 per head, pure VALU over LDS ----
    if (tid < TB * NHEAD) {
        const int bi = tid >> 3, hh = tid & 7;
        const bf16* qb = QKV + (bi * 3) * LQKV + hh * HDIM;
        const bf16* kb = qb + DMODEL;
        const bf16* vb = qb + 2 * DMODEL;
        float sc[3][3];
#pragma unroll
        for (int i = 0; i < 3; ++i)
#pragma unroll
            for (int j = 0; j < 3; ++j) {
                float d = 0.f;
                for (int e = 0; e < HDIM; ++e)
                    d += (float)qb[i * LQKV + e] * (float)kb[j * LQKV + e];
                sc[i][j] = d * 0.125f;                    // 1/sqrt(64)
            }
#pragma unroll
        for (int i = 0; i < 3; ++i) {
            float mx = fmaxf(sc[i][0], fmaxf(sc[i][1], sc[i][2]));
            float e0 = __expf(sc[i][0] - mx), e1 = __expf(sc[i][1] - mx), e2 = __expf(sc[i][2] - mx);
            float inv = 1.f / (e0 + e1 + e2);
            sc[i][0] = e0 * inv; sc[i][1] = e1 * inv; sc[i][2] = e2 * inv;
        }
#pragma unroll
        for (int i = 0; i < 3; ++i)
            for (int e = 0; e < HDIM; ++e) {
                float o = sc[i][0] * (float)vb[0 * LQKV + e]
                        + sc[i][1] * (float)vb[1 * LQKV + e]
                        + sc[i][2] * (float)vb[2 * LQKV + e];
                AO[(bi * 3 + i) * LXB + hh * HDIM + e] = (bf16)o;
            }
    }
    __syncthreads();

    // ---- 4) out-proj + residual -> f32 buffer (3 M-tiles per N-column) ----
    for (int nt = wave; nt < 32; nt += NWAVES) {
        v8f acc[3];
#pragma unroll
        for (int m = 0; m < 3; ++m) acc[m] = (v8f){0.f,0.f,0.f,0.f,0.f,0.f,0.f,0.f};
        for (int kt = 0; kt < DMODEL / 32; ++kt) {
            v16bf bw = load_b(wout_f, nt * (DMODEL / 32) + kt, lane);
#pragma unroll
            for (int m = 0; m < 3; ++m) {
                v16bf a = load_a(AO + m * 16 * LXB, LXB, kt * 32, lane);
                acc[m] = wmma_bf16(a, bw, acc[m]);
            }
        }
        const int col = nt * 16 + ln15;
        const float bias = b_out[col];
#pragma unroll
        for (int m = 0; m < 3; ++m) {
            const int mbase = m * 16 + mb8;
#pragma unroll
            for (int r = 0; r < 8; ++r) {
                const int row = mbase + r;
                X2F[row * LX2 + col] = acc[m][r] + bias + (float)XB[row * LXB + col];
            }
        }
    }
    __syncthreads();

    // ---- 5) LayerNorm 1 -> x1 (bf16, overwrites XB) ----
    if (tid < MROWS) {
        float s = 0.f, s2 = 0.f;
        for (int c = 0; c < DMODEL; ++c) {
            float v = X2F[tid * LX2 + c];
            s += v; s2 += v * v;
        }
        const float mu  = s * (1.f / DMODEL);
        const float var = s2 * (1.f / DMODEL) - mu * mu;
        MU[tid] = mu; RS[tid] = rsqrtf(var + 1e-5f);
    }
    __syncthreads();
    for (int idx = tid; idx < MROWS * DMODEL; idx += NTHREADS) {
        const int r = idx >> 9, c = idx & (DMODEL - 1);
        float v = (X2F[r * LX2 + c] - MU[r]) * RS[r] * ln1_g[c] + ln1_b[c];
        XB[r * LXB + c] = (bf16)v;
    }
    __syncthreads();

    // ---- 6) FFN: 8 chunks of 256 FF dims; output tiles pinned in VGPRs ----
    // wave owns 4 N-columns of the D-output: nt = wave + j*8, j=0..3; x 3 M-tiles.
    v8f accff[12];                         // [m*4 + j]
#pragma unroll
    for (int i = 0; i < 12; ++i) accff[i] = (v8f){0.f,0.f,0.f,0.f,0.f,0.f,0.f,0.f};

    for (int c = 0; c < FFDIM / 256; ++c) {
        // GEMM1 chunk: h = relu(x1 * w1^T), 3 M-tiles per N-column
        for (int nt = wave; nt < 16; nt += NWAVES) {
            const int ntg = c * 16 + nt;       // global ntile into w1
            v8f acc[3];
#pragma unroll
            for (int m = 0; m < 3; ++m) acc[m] = (v8f){0.f,0.f,0.f,0.f,0.f,0.f,0.f,0.f};
            for (int kt = 0; kt < DMODEL / 32; ++kt) {
                v16bf bw = load_b(w1_f, ntg * (DMODEL / 32) + kt, lane);
#pragma unroll
                for (int m = 0; m < 3; ++m) {
                    v16bf a = load_a(XB + m * 16 * LXB, LXB, kt * 32, lane);
                    acc[m] = wmma_bf16(a, bw, acc[m]);
                }
            }
            const int coll = nt * 16 + ln15;
            const float bias = b1[c * 256 + coll];
#pragma unroll
            for (int m = 0; m < 3; ++m) {
                const int mbase = m * 16 + mb8;
#pragma unroll
                for (int r = 0; r < 8; ++r)
                    H[(mbase + r) * LH + coll] = (bf16)fmaxf(acc[m][r] + bias, 0.f);
            }
        }
        __syncthreads();
        // GEMM2 partial: one B fragment feeds 3 M-tile accumulations
        for (int kt = 0; kt < 8; ++kt) {       // 256 / 32
            v16bf a0 = load_a(H + 0 * 16 * LH, LH, kt * 32, lane);
            v16bf a1 = load_a(H + 1 * 16 * LH, LH, kt * 32, lane);
            v16bf a2 = load_a(H + 2 * 16 * LH, LH, kt * 32, lane);
#pragma unroll
            for (int j = 0; j < 4; ++j) {
                const int nt = wave + j * NWAVES;
                v16bf bw = load_b(w2_f, nt * (FFDIM / 32) + c * 8 + kt, lane);
                accff[0 * 4 + j] = wmma_bf16(a0, bw, accff[0 * 4 + j]);
                accff[1 * 4 + j] = wmma_bf16(a1, bw, accff[1 * 4 + j]);
                accff[2 * 4 + j] = wmma_bf16(a2, bw, accff[2 * 4 + j]);
            }
        }
        __syncthreads();
    }

    // ---- 7) FFN bias + residual -> f32 buffer ----
#pragma unroll
    for (int j = 0; j < 4; ++j) {
        const int nt = wave + j * NWAVES;
        const int col = nt * 16 + ln15;
        const float bias = b2[col];
#pragma unroll
        for (int m = 0; m < 3; ++m) {
            const int mbase = m * 16 + mb8;
#pragma unroll
            for (int r = 0; r < 8; ++r) {
                const int row = mbase + r;
                X2F[row * LX2 + col] = accff[m * 4 + j][r] + bias + (float)XB[row * LXB + col];
            }
        }
    }
    __syncthreads();

    // ---- 8) LayerNorm 2 + mean over seq -> output ----
    if (tid < MROWS) {
        float s = 0.f, s2 = 0.f;
        for (int c = 0; c < DMODEL; ++c) {
            float v = X2F[tid * LX2 + c];
            s += v; s2 += v * v;
        }
        const float mu  = s * (1.f / DMODEL);
        const float var = s2 * (1.f / DMODEL) - mu * mu;
        MU[tid] = mu; RS[tid] = rsqrtf(var + 1e-5f);
    }
    __syncthreads();
    for (int idx = tid; idx < TB * DMODEL; idx += NTHREADS) {
        const int bi = idx >> 9, d = idx & (DMODEL - 1);
        const float g = ln2_g[d], bb = ln2_b[d];
        float s = 0.f;
#pragma unroll
        for (int q = 0; q < 3; ++q) {
            const int r = bi * 3 + q;
            s += (X2F[r * LX2 + d] - MU[r]) * RS[r] * g + bb;
        }
        out[(size_t)(b0 + bi) * DMODEL + d] = s * (1.f / 3.f);
    }
}

extern "C" void kernel_launch(void* const* d_in, const int* in_sizes, int n_in,
                              void* d_out, int out_size, void* d_ws, size_t ws_size,
                              hipStream_t stream) {
    (void)in_sizes; (void)n_in; (void)out_size; (void)ws_size;
    const float* feat0   = (const float*)d_in[0];
    const float* feat1   = (const float*)d_in[1];
    const float* feat2   = (const float*)d_in[2];
    const float* pos_emb = (const float*)d_in[3];
    const float* w_in    = (const float*)d_in[4];
    const float* b_in    = (const float*)d_in[5];
    const float* w_out   = (const float*)d_in[6];
    const float* b_out   = (const float*)d_in[7];
    const float* ln1_g   = (const float*)d_in[8];
    const float* ln1_b   = (const float*)d_in[9];
    const float* w1      = (const float*)d_in[10];
    const float* b1      = (const float*)d_in[11];
    const float* w2      = (const float*)d_in[12];
    const float* b2      = (const float*)d_in[13];
    const float* ln2_g   = (const float*)d_in[14];
    const float* ln2_b   = (const float*)d_in[15];
    float* out = (float*)d_out;

    char* ws = (char*)d_ws;
    bf16* win_f  = (bf16*)(ws + WS_WIN);
    bf16* wout_f = (bf16*)(ws + WS_WOUT);
    bf16* w1_f   = (bf16*)(ws + WS_W1);
    bf16* w2_f   = (bf16*)(ws + WS_W2);

    // ---- prepass: repack all weights to bf16 fragment-major (6 MB total) ----
    {
        const int nfrag_win  = (3 * DMODEL / 16) * (DMODEL / 32);   // 1536
        const int nfrag_wout = (DMODEL / 16) * (DMODEL / 32);       // 512
        const int nfrag_w1   = (FFDIM / 16) * (DMODEL / 32);        // 2048
        const int nfrag_w2   = (DMODEL / 16) * (FFDIM / 32);        // 2048
        repack_weight<<<(nfrag_win  * 32 + NTHREADS - 1) / NTHREADS, NTHREADS, 0, stream>>>(w_in,  win_f,  3 * DMODEL, DMODEL);
        repack_weight<<<(nfrag_wout * 32 + NTHREADS - 1) / NTHREADS, NTHREADS, 0, stream>>>(w_out, wout_f, DMODEL,     DMODEL);
        repack_weight<<<(nfrag_w1   * 32 + NTHREADS - 1) / NTHREADS, NTHREADS, 0, stream>>>(w1,    w1_f,   FFDIM,      DMODEL);
        repack_weight<<<(nfrag_w2   * 32 + NTHREADS - 1) / NTHREADS, NTHREADS, 0, stream>>>(w2,    w2_f,   DMODEL,     FFDIM);
    }

    // Opt in to the large (>64KB default) dynamic LDS allocation (WGP has 320KB).
    (void)hipFuncSetAttribute((const void*)fused_transformer_block,
                              hipFuncAttributeMaxDynamicSharedMemorySize, SMEM_BYTES);

    fused_transformer_block<<<BATCH / TB, NTHREADS, SMEM_BYTES, stream>>>(
        feat0, feat1, feat2, pos_emb, win_f, b_in, wout_f, b_out,
        ln1_g, ln1_b, w1_f, b1, w2_f, b2, ln2_g, ln2_b, out);
}
